// ICMDynamics_64501818851840
// MI455X (gfx1250) — compile-verified, matrix operand order checked by reading, not run on hardware
//
#include <hip/hip_runtime.h>
#include <hip/hip_bf16.h>

// ---------------------------------------------------------------------------
// ICM dynamics (forward + inverse) fused kernel for gfx1250 (MI455X).
// f16 multiply / f32 accumulate via V_WMMA_F32_16X16X32_F16.
// ---------------------------------------------------------------------------

typedef _Float16 v16h __attribute__((ext_vector_type(16)));
typedef _Float16 v8h  __attribute__((ext_vector_type(8)));
typedef _Float16 h4   __attribute__((ext_vector_type(4)));
typedef float    v8f  __attribute__((ext_vector_type(8)));

#define F_FRAMES 4096
#define N_AGENTS 64
#define D_FEAT   256
#define HID      140
#define A_TOTAL  24

// padded GEMM geometry
#define K1 288   // 280 -> 9 k-tiles of 32   (fwd_in)
#define N1 144   // 140 -> 9 n-tiles of 16   (hidden)
#define K2 160   // 140 -> 5 k-tiles of 32   (hidden as K)
#define N2 256   // 16 n-tiles               (pred)
#define K3 512   // 16 k-tiles               (inv_in)
#define N3 32    // 24 -> 2 n-tiles          (logits)

#define W1_FRAGS (9 * 9)    // 81
#define W2_FRAGS (5 * 16)   // 80
#define WI_FRAGS (16 * 2)   // 32
#define FRAG_HALVES 512     // 32x16 f16 fragment

// LDS row strides (halves), multiples of 8 so 16B chunks stay aligned
#define SA1 296   // 288 + 8
#define SA2 264   // 256 + 8
#define SH  168   // 160 + 8

// ---------------------------------------------------------------------------
// Prepack: f32 weight [Kreal x Nreal] row-major -> f16 WMMA B fragments.
// Fragment (kt,nt): lane l holds 16 halves; half idx -> K per CDNA5 16-bit
// B layout: k = kt*32 + (idx<8 ? idx : idx+8) + (lane<16 ? 0 : 8), n = nt*16
// + (lane&15). Out-of-range (K/N padding) entries written as zero.
// ---------------------------------------------------------------------------
__global__ void icm_prepack(const float* __restrict__ W, int Kreal, int Nreal,
                            int KT, int NT, _Float16* __restrict__ dst) {
    int q = blockIdx.x * blockDim.x + threadIdx.x;
    int total = KT * NT * FRAG_HALVES;
    if (q >= total) return;
    int frag = q >> 9;
    int r    = q & 511;
    int lane = r >> 4;
    int idx  = r & 15;
    int kt = frag / NT;
    int nt = frag - kt * NT;
    int k = kt * 32 + ((idx < 8) ? idx : idx + 8) + ((lane < 16) ? 0 : 8);
    int n = nt * 16 + (lane & 15);
    float v = (k < Kreal && n < Nreal) ? W[(long)k * Nreal + n] : 0.0f;
    dst[q] = (_Float16)v;
}

// ---------------------------------------------------------------------------
// Main fused kernel: one block = one frame (64 rows).
// ---------------------------------------------------------------------------
__global__ __launch_bounds__(256) void icm_main(
    const float* __restrict__ cur, const float* __restrict__ nxt,
    const int* __restrict__ actions,
    const float* __restrict__ b1, const float* __restrict__ b2,
    const float* __restrict__ bi,
    const _Float16* __restrict__ W1p, const _Float16* __restrict__ W2p,
    const _Float16* __restrict__ Wip,
    float* __restrict__ out1, float* __restrict__ out2) {

    __shared__ _Float16 sA1[64 * SA1];   // fwd_in / inv_in first half (f16)
    __shared__ _Float16 sA2[64 * SA2];   // next_feature (f16)
    __shared__ _Float16 sHH[64 * SH];    // hidden activations (f16)

    const int tid = threadIdx.x;
    const int fblk = blockIdx.x;
    const long rowBase = (long)fblk * 64;

    // ---- phase A: stream features in (f32 -> f16), zero pads -------------
    for (int q = tid; q < 64 * 64; q += 256) {          // 64 rows x 64 float4
        int rw = q >> 6, c4 = q & 63;
        float4 v = ((const float4*)(cur + (rowBase + rw) * D_FEAT))[c4];
        h4 hv = { (_Float16)v.x, (_Float16)v.y, (_Float16)v.z, (_Float16)v.w };
        *(h4*)&sA1[rw * SA1 + c4 * 4] = hv;
    }
    for (int q = tid; q < 64 * 64; q += 256) {
        int rw = q >> 6, c4 = q & 63;
        float4 v = ((const float4*)(nxt + (rowBase + rw) * D_FEAT))[c4];
        h4 hv = { (_Float16)v.x, (_Float16)v.y, (_Float16)v.z, (_Float16)v.w };
        *(h4*)&sA2[rw * SA2 + c4 * 4] = hv;
    }
    // zero one-hot tail of A1 (cols 256..287)
    for (int q = tid; q < 64 * 8; q += 256) {           // 8 h4 per row
        int rw = q >> 3, c4 = q & 7;
        h4 z = { (_Float16)0.f, (_Float16)0.f, (_Float16)0.f, (_Float16)0.f };
        *(h4*)&sA1[rw * SA1 + 256 + c4 * 4] = z;
    }
    // zero hidden buffer (cols 140..159 must stay 0 for GEMM2 K-padding)
    for (int q = tid; q < 64 * (SH / 4); q += 256) {
        int rw = q / (SH / 4), c4 = q - rw * (SH / 4);
        h4 z = { (_Float16)0.f, (_Float16)0.f, (_Float16)0.f, (_Float16)0.f };
        *(h4*)&sHH[rw * SH + c4 * 4] = z;
    }
    __syncthreads();

    // ---- phase B: scatter one-hot actions into A1 tail -------------------
    if (tid < 64 * 3) {
        int rw = tid / 3, t = tid - rw * 3;
        int a = actions[(long)fblk * 3 * N_AGENTS + t * N_AGENTS + rw]; // 0..7
        sA1[rw * SA1 + 256 + t * 8 + a] = (_Float16)1.0f;
    }
    __syncthreads();

    // ---- compute ---------------------------------------------------------
    const int wave  = tid >> 5;
    const int lane  = tid & 31;
    const int ml    = lane & 15;
    const int khalf = (lane < 16) ? 0 : 8;

    // A fragment: per-lane two aligned 16B chunks from row-major f16 LDS tile
    auto loadA = [&](const _Float16* base, int stride, int mrow,
                     int kbase) -> v16h {
        const _Float16* p = base + mrow * stride + kbase;
        v8h lo = *(const v8h*)p;
        v8h hi = *(const v8h*)(p + 16);
        return __builtin_shufflevector(lo, hi, 0, 1, 2, 3, 4, 5, 6, 7, 8, 9,
                                       10, 11, 12, 13, 14, 15);
    };
    // B fragment: per-lane 32 contiguous bytes of prepacked weights (L2-hot)
    auto loadB = [&](const _Float16* base, int frag) -> v16h {
        const _Float16* p = base + frag * FRAG_HALVES + lane * 16;
        v8h lo = *(const v8h*)p;
        v8h hi = *(const v8h*)(p + 8);
        return __builtin_shufflevector(lo, hi, 0, 1, 2, 3, 4, 5, 6, 7, 8, 9,
                                       10, 11, 12, 13, 14, 15);
    };

    // ---- GEMM3: inverse dynamics, [64x512] x [512x32] -> out2 ------------
    {
        int t = wave;                 // 8 tiles, 8 waves
        int mt = t >> 1, nt = t & 1;
        v8f acc = {};
#pragma unroll
        for (int kt = 0; kt < 8; ++kt) {
            v16h a = loadA(sA1, SA1, mt * 16 + ml, kt * 32 + khalf);
            v16h b = loadB(Wip, kt * 2 + nt);
            acc = __builtin_amdgcn_wmma_f32_16x16x32_f16(false, a, false, b,
                                                         (short)0, acc, false,
                                                         false);
        }
#pragma unroll
        for (int kt = 0; kt < 8; ++kt) {
            v16h a = loadA(sA2, SA2, mt * 16 + ml, kt * 32 + khalf);
            v16h b = loadB(Wip, (kt + 8) * 2 + nt);
            acc = __builtin_amdgcn_wmma_f32_16x16x32_f16(false, a, false, b,
                                                         (short)0, acc, false,
                                                         false);
        }
        int col = nt * 16 + ml;
        if (col < A_TOTAL) {
            float bv = bi[col];
#pragma unroll
            for (int v = 0; v < 8; ++v) {
                int rowm = mt * 16 + ((lane < 16) ? v : v + 8);
                out2[(rowBase + rowm) * A_TOTAL + col] = acc[v] + bv;
            }
        }
    }

    // ---- GEMM1: forward hidden, [64x288] x [288x144] -> leaky -> sHH -----
    for (int t = wave; t < 4 * 9; t += 8) {
        int mt = t / 9, nt = t - mt * 9;
        v8f acc = {};
#pragma unroll
        for (int kt = 0; kt < 9; ++kt) {
            v16h a = loadA(sA1, SA1, mt * 16 + ml, kt * 32 + khalf);
            v16h b = loadB(W1p, kt * 9 + nt);
            acc = __builtin_amdgcn_wmma_f32_16x16x32_f16(false, a, false, b,
                                                         (short)0, acc, false,
                                                         false);
        }
        int col = nt * 16 + ml;
        if (col < HID) {
            float bv = b1[col];
#pragma unroll
            for (int v = 0; v < 8; ++v) {
                float x = acc[v] + bv;
                x = (x > 0.f) ? x : 0.1f * x;          // leaky_relu(0.1)
                int rowm = mt * 16 + ((lane < 16) ? v : v + 8);
                sHH[rowm * SH + col] = (_Float16)x;
            }
        }
    }
    __syncthreads();

    // ---- GEMM2: prediction, [64x160] x [160x256] -> out1 -----------------
    for (int t = wave; t < 4 * 16; t += 8) {
        int mt = t >> 4, nt = t & 15;
        v8f acc = {};
#pragma unroll
        for (int kt = 0; kt < 5; ++kt) {
            v16h a = loadA(sHH, SH, mt * 16 + ml, kt * 32 + khalf);
            v16h b = loadB(W2p, kt * 16 + nt);
            acc = __builtin_amdgcn_wmma_f32_16x16x32_f16(false, a, false, b,
                                                         (short)0, acc, false,
                                                         false);
        }
        int col = nt * 16 + ml;
        float bv = b2[col];
#pragma unroll
        for (int v = 0; v < 8; ++v) {
            int rowm = mt * 16 + ((lane < 16) ? v : v + 8);
            out1[(rowBase + rowm) * D_FEAT + col] = acc[v] + bv;
        }
    }
}

// ---------------------------------------------------------------------------
extern "C" void kernel_launch(void* const* d_in, const int* in_sizes, int n_in,
                              void* d_out, int out_size, void* d_ws,
                              size_t ws_size, hipStream_t stream) {
    const float* cur     = (const float*)d_in[0];
    const float* nxt     = (const float*)d_in[1];
    const int*   actions = (const int*)d_in[2];
    const float* W1      = (const float*)d_in[3];
    const float* b1      = (const float*)d_in[4];
    const float* W2      = (const float*)d_in[5];
    const float* b2      = (const float*)d_in[6];
    const float* Wi      = (const float*)d_in[7];
    const float* bi      = (const float*)d_in[8];

    float* out1 = (float*)d_out;                                   // [M,256]
    float* out2 = out1 + (size_t)F_FRAMES * N_AGENTS * D_FEAT;     // [M,24]

    _Float16* wsH = (_Float16*)d_ws;
    _Float16* W1p = wsH;                                   // 81*512 halves
    _Float16* W2p = W1p + W1_FRAGS * FRAG_HALVES;          // 80*512 halves
    _Float16* Wip = W2p + W2_FRAGS * FRAG_HALVES;          // 32*512 halves

    // prepack weights into WMMA B-fragment layout (runs every call; cheap,
    // deterministic, graph-capture safe)
    {
        int t1 = W1_FRAGS * FRAG_HALVES;
        int t2 = W2_FRAGS * FRAG_HALVES;
        int t3 = WI_FRAGS * FRAG_HALVES;
        icm_prepack<<<(t1 + 255) / 256, 256, 0, stream>>>(W1, 280, HID, 9, 9,
                                                          W1p);
        icm_prepack<<<(t2 + 255) / 256, 256, 0, stream>>>(W2, HID, D_FEAT, 5,
                                                          16, W2p);
        icm_prepack<<<(t3 + 255) / 256, 256, 0, stream>>>(Wi, 2 * D_FEAT,
                                                          A_TOTAL, 16, 2, Wip);
    }

    icm_main<<<F_FRAMES, 256, 0, stream>>>(cur, nxt, actions, b1, b2, bi, W1p,
                                           W2p, Wip, out1, out2);
}
